// EncoderBlock_61194694034334
// MI455X (gfx1250) — matrix operand order repaired and checked by reading, not compiled, not run on hardware
//
#include <hip/hip_runtime.h>
#include <hip/hip_bf16.h>
#include <math.h>

// ---------------------------------------------------------------------------
// Transformer encoder block for MI455X (gfx1250, wave32, WMMA 16x16x32 f16).
// Flash-attention avoids the 2GB score tensor; all GEMMs use
// v_wmma_f32_16x16x32_f16 with f32 accumulation. Attention K/V tiles are
// staged block-wide into LDS with async global->LDS copies (ASYNCcnt),
// double-buffered so the DMA for the next 64-key block overlaps compute.
// ---------------------------------------------------------------------------

typedef __attribute__((ext_vector_type(16))) _Float16 v16h;
typedef __attribute__((ext_vector_type(8)))  _Float16 v8h;
typedef __attribute__((ext_vector_type(8)))  float    v8f;

union AFrag { v16h v; v8h h[2]; };

__device__ inline v8f wmma_f16(v16h a, v16h b, v8f c) {
    return __builtin_amdgcn_wmma_f32_16x16x32_f16(
        /*neg_a=*/false, a, /*neg_b=*/false, b,
        /*c_mod=*/(short)0, c, /*reuse_a=*/false, /*reuse_b=*/false);
}

// Async global -> LDS copy, 16 bytes per lane. lds_off is the workgroup-
// relative LDS byte offset (low 32 bits of a generic LDS pointer), gptr a
// per-lane global address. Tracked by ASYNCcnt.
__device__ inline void async_copy_b128(unsigned lds_off, const void* gptr) {
    asm volatile("global_load_async_to_lds_b128 %0, %1, off"
                 :: "v"(lds_off), "v"(gptr) : "memory");
}
__device__ inline void wait_asynccnt0() {
    asm volatile("s_wait_asynccnt 0" ::: "memory");
}

#define BB 2
#define NN 4096
#define DD 256
#define HH 8
#define HD 32
#define DFF 1024

// ---------------------------------------------------------------------------
// Weight convert + transpose: W[Din,Dout] f32 -> Wt[Dout,Din] f16
// ---------------------------------------------------------------------------
__global__ void cvt_t_kernel(const float* __restrict__ W, _Float16* __restrict__ Wt,
                             int Din, int Dout) {
    int idx = blockIdx.x * blockDim.x + threadIdx.x;
    if (idx >= Din * Dout) return;
    int in = idx / Dout, o = idx % Dout;
    Wt[(size_t)o * Din + in] = (_Float16)W[idx];
}

// ---------------------------------------------------------------------------
// LayerNorm matching reference: (x - mean) / (std + 1e-6), writes f16
// One 256-thread block per row, Dim == 256.
// ---------------------------------------------------------------------------
__global__ __launch_bounds__(256) void layernorm_kernel(
    const float* __restrict__ x, _Float16* __restrict__ out) {
    __shared__ float red[256];
    const int row = blockIdx.x;
    const int tid = threadIdx.x;
    float v = x[(size_t)row * DD + tid];
    red[tid] = v; __syncthreads();
    for (int s = 128; s > 0; s >>= 1) { if (tid < s) red[tid] += red[tid + s]; __syncthreads(); }
    float mean = red[0] * (1.0f / DD);
    __syncthreads();
    float d = v - mean;
    red[tid] = d * d; __syncthreads();
    for (int s = 128; s > 0; s >>= 1) { if (tid < s) red[tid] += red[tid + s]; __syncthreads(); }
    float var = red[0] * (1.0f / DD);
    float denom = sqrtf(var) + 1e-6f;
    out[(size_t)row * DD + tid] = (_Float16)(d / denom);
}

// ---------------------------------------------------------------------------
// Generic WMMA GEMM: C[M,Nout] = A[M,K](f16) @ Bt[Nout,K](f16)^T + bias
// 128 threads = 4 waves; each wave: 16 rows x 64 cols.
// Epilogue modes:
//   0: Q/K head-major f16  out[((b*H+h)*N + n)*32 + hd]
//   1: V transposed  f16   out[((b*H+h)*32 + hd)*N + n]
//   2: f32 out = acc + bias + res   (attn@Wo + bo + x -> x_skip)
//   3: f16 gelu(acc + bias)         (MLP hidden)
//   4: f32 out = acc + bias + res   (final: + x_skip)
// ---------------------------------------------------------------------------
__global__ __launch_bounds__(128) void gemm_f16_kernel(
    const _Float16* __restrict__ A, const _Float16* __restrict__ Bt,
    const float* __restrict__ bias, const float* __restrict__ res,
    void* __restrict__ out, int M, int K, int Nout, int mode) {
    const int lane = threadIdx.x & 31;
    const int wave = threadIdx.x >> 5;
    const int cl   = lane & 15;
    const int half = lane >> 4;
    const int blockM = blockIdx.y * 64;
    const int blockN = blockIdx.x * 64;
    const int row   = blockM + wave * 16 + cl;
    const int akoff = half * 16;
    const int bkoff = half * 16;

    v8f acc[4] = {};
    for (int kk = 0; kk < K; kk += 32) {
        AFrag a;
        const _Float16* ap = A + (size_t)row * K + kk + akoff;
        a.h[0] = *(const v8h*)(ap);
        a.h[1] = *(const v8h*)(ap + 8);
#pragma unroll
        for (int t = 0; t < 4; ++t) {
            int col = blockN + t * 16 + cl;
            v16h b = *(const v16h*)(Bt + (size_t)col * K + kk + bkoff);
            acc[t] = wmma_f16(a.v, b, acc[t]);
        }
    }

#pragma unroll
    for (int t = 0; t < 4; ++t) {
        const int c  = blockN + t * 16 + cl;
        const float bv = bias[c];
#pragma unroll
        for (int r = 0; r < 8; ++r) {
            const int m = blockM + wave * 16 + r + 8 * half;
            float v = acc[t][r] + bv;
            if (mode == 0) {
                int b = m >> 12, n = m & (NN - 1);
                int h = c >> 5, hd = c & 31;
                ((_Float16*)out)[(((size_t)(b * HH + h) * NN + n) << 5) + hd] = (_Float16)v;
            } else if (mode == 1) {
                int b = m >> 12, n = m & (NN - 1);
                int h = c >> 5, hd = c & 31;
                ((_Float16*)out)[((size_t)(b * HH + h) * HD + hd) * NN + n] = (_Float16)v;
            } else if (mode == 2) {
                ((float*)out)[(size_t)m * Nout + c] = v + res[(size_t)m * Nout + c];
            } else if (mode == 3) {
                float u = v;
                float g = 0.5f * u * (1.0f + tanhf(0.7978845608028654f * (u + 0.044715f * u * u * u)));
                ((_Float16*)out)[(size_t)m * Nout + c] = (_Float16)g;
            } else {
                ((float*)out)[(size_t)m * Nout + c] = v + res[(size_t)m * Nout + c];
            }
        }
    }
}

// ---------------------------------------------------------------------------
// Flash attention. Q,K in [B*H, N, 32] f16 ; V transposed [B*H, 32, N] f16.
// Block = 4 waves, each wave owns a 16-query tile; the block shares 64-key
// K/V tiles staged into LDS via async global->LDS copies, double-buffered.
// Per 64-key block and wave: 4 score WMMAs (K = head dim = 32), one online
// softmax pass (16-lane shfl reductions), P bounced via wave-private LDS
// (C layout -> A layout), 4 PV WMMAs.  grid = (B*H, N/64), block = 128.
// ---------------------------------------------------------------------------
__global__ __launch_bounds__(128) void flash_attn_kernel(
    const _Float16* __restrict__ Q, const _Float16* __restrict__ Kmat,
    const _Float16* __restrict__ Vt, _Float16* __restrict__ attn_out) {
    __shared__ __align__(64) _Float16 Kl[2][64][32];   // [buf][key][hd]
    __shared__ __align__(64) _Float16 Vl[2][32][64];   // [buf][hd][key]
    __shared__ __align__(64) _Float16 Ptile[4][16][64];
    const int tid  = threadIdx.x;
    const int lane = tid & 31;
    const int wave = tid >> 5;
    const int cl   = lane & 15;
    const int half = lane >> 4;
    const int bh = blockIdx.x;
    const int b = bh >> 3, h = bh & 7;
    const int q0 = (blockIdx.y * 4 + wave) * 16;

    const _Float16* Qh = Q    + (size_t)bh * NN * HD;
    const _Float16* Kh = Kmat + (size_t)bh * NN * HD;
    const _Float16* Vh = Vt   + (size_t)bh * HD * NN;

    // Stage one 64-key K/V tile into LDS buffer `buf` (block-wide, async).
    // K tile: keys kb..kb+63 are one contiguous 4KB chunk of Kh.
    // V tile: 32 rows (hd) of 128B each from Vh.
    auto stage = [&](int buf, int kb) {
        const char* kg = (const char*)(Kh + (size_t)kb * HD);
        unsigned kofs = (unsigned)(size_t)&Kl[buf][0][0];
#pragma unroll
        for (int i = 0; i < 2; ++i) {
            int c = i * 128 + tid;                       // 16B chunk id, 0..255
            async_copy_b128(kofs + c * 16, kg + c * 16);
        }
        unsigned vofs = (unsigned)(size_t)&Vl[buf][0][0];
#pragma unroll
        for (int i = 0; i < 2; ++i) {
            int c = i * 128 + tid;
            int hd = c >> 3, part = c & 7;               // 8 chunks per 128B row
            const char* vg = (const char*)(Vh + (size_t)hd * NN + kb) + part * 16;
            async_copy_b128(vofs + hd * 128 + part * 16, vg);
        }
    };

    AFrag qf;
    {
        const _Float16* qp = Qh + (size_t)(q0 + cl) * HD + half * 16;
        qf.h[0] = *(const v8h*)(qp);
        qf.h[1] = *(const v8h*)(qp + 8);
    }

    float m_i[8], l_i[8];
#pragma unroll
    for (int r = 0; r < 8; ++r) { m_i[r] = -1e30f; l_i[r] = 0.0f; }
    v8f O0 = {}, O1 = {};
    const float scale = 0.17677669529663687f; // 1/sqrt(32)

    stage(0, 0);
    for (int kb = 0; kb < NN; kb += 64) {
        const int buf = (kb >> 6) & 1;
        wait_asynccnt0();       // this wave's copies for `buf` have landed
        __syncthreads();        // => everyone's copies for `buf` have landed
        if (kb + 64 < NN) stage(buf ^ 1, kb + 64);   // overlap next DMA

        // ---- scores: 4 x 16-key subtiles, K-dim = head dim = 32 ----
        v8f S[4];
        v8f z = {};
#pragma unroll
        for (int t = 0; t < 4; ++t) {
            v16h kf = *(const v16h*)&Kl[buf][t * 16 + cl][half * 16];
            S[t] = wmma_f16(qf.v, kf, z);
        }

        // ---- online softmax across the 64-key block ----
#pragma unroll
        for (int r = 0; r < 8; ++r) {
            float s0 = S[0][r] * scale, s1 = S[1][r] * scale;
            float s2 = S[2][r] * scale, s3 = S[3][r] * scale;
            float mx = fmaxf(fmaxf(s0, s1), fmaxf(s2, s3));
            for (int off = 1; off < 16; off <<= 1)
                mx = fmaxf(mx, __shfl_xor(mx, off, 32));
            float mnew  = fmaxf(m_i[r], mx);
            float alpha = __expf(m_i[r] - mnew);
            float p0 = __expf(s0 - mnew), p1 = __expf(s1 - mnew);
            float p2 = __expf(s2 - mnew), p3 = __expf(s3 - mnew);
            float rs = (p0 + p1) + (p2 + p3);
            for (int off = 1; off < 16; off <<= 1)
                rs += __shfl_xor(rs, off, 32);
            l_i[r] = l_i[r] * alpha + rs;
            m_i[r] = mnew;
            O0[r] *= alpha;
            O1[r] *= alpha;
            int prow = r + 8 * half;
            Ptile[wave][prow][cl]      = (_Float16)p0;
            Ptile[wave][prow][16 + cl] = (_Float16)p1;
            Ptile[wave][prow][32 + cl] = (_Float16)p2;
            Ptile[wave][prow][48 + cl] = (_Float16)p3;
        }

        // ---- O += P @ V : two K=32 steps over the 64-key block ----
#pragma unroll
        for (int s2 = 0; s2 < 2; ++s2) {
            AFrag pf;
            pf.h[0] = *(const v8h*)&Ptile[wave][cl][s2 * 32 + half * 16];
            pf.h[1] = *(const v8h*)&Ptile[wave][cl][s2 * 32 + half * 16 + 8];
            v16h v0 = *(const v16h*)&Vl[buf][cl][s2 * 32 + half * 16];
            v16h v1 = *(const v16h*)&Vl[buf][16 + cl][s2 * 32 + half * 16];
            O0 = wmma_f16(pf.v, v0, O0);
            O1 = wmma_f16(pf.v, v1, O1);
        }
    }

#pragma unroll
    for (int r = 0; r < 8; ++r) {
        float inv = 1.0f / l_i[r];
        int qrow = q0 + r + 8 * half;
        size_t base = ((size_t)(b * NN + qrow)) * DD + h * HD;
        attn_out[base + cl]      = (_Float16)(O0[r] * inv);
        attn_out[base + 16 + cl] = (_Float16)(O1[r] * inv);
    }
}

// ---------------------------------------------------------------------------
extern "C" void kernel_launch(void* const* d_in, const int* in_sizes, int n_in,
                              void* d_out, int out_size, void* d_ws, size_t ws_size,
                              hipStream_t stream) {
    const float* x  = (const float*)d_in[0];
    const float* Wq = (const float*)d_in[1];
    const float* bq = (const float*)d_in[2];
    const float* Wk = (const float*)d_in[3];
    const float* bk = (const float*)d_in[4];
    const float* Wv = (const float*)d_in[5];
    const float* bv = (const float*)d_in[6];
    const float* Wo = (const float*)d_in[7];
    const float* bo = (const float*)d_in[8];
    const float* W1 = (const float*)d_in[9];
    const float* b1 = (const float*)d_in[10];
    const float* W2 = (const float*)d_in[11];
    const float* b2 = (const float*)d_in[12];
    float* out = (float*)d_out;

    char* wsp = (char*)d_ws;
    auto alloc = [&](size_t bytes) -> char* {
        char* p = wsp;
        wsp += (bytes + 255) & ~(size_t)255;
        return p;
    };
    const size_t M = (size_t)BB * NN; // 8192
    _Float16* Wq_t = (_Float16*)alloc((size_t)DD * DD * 2);
    _Float16* Wk_t = (_Float16*)alloc((size_t)DD * DD * 2);
    _Float16* Wv_t = (_Float16*)alloc((size_t)DD * DD * 2);
    _Float16* Wo_t = (_Float16*)alloc((size_t)DD * DD * 2);
    _Float16* W1_t = (_Float16*)alloc((size_t)DD * DFF * 2);
    _Float16* W2_t = (_Float16*)alloc((size_t)DFF * DD * 2);
    _Float16* xn   = (_Float16*)alloc(M * DD * 2);   // xn1, later reused as xn2
    _Float16* qb   = (_Float16*)alloc(M * DD * 2);
    _Float16* kb   = (_Float16*)alloc(M * DD * 2);
    _Float16* vtb  = (_Float16*)alloc(M * DD * 2);
    _Float16* attn = (_Float16*)alloc(M * DD * 2);
    float*    xsk  = (float*)   alloc(M * DD * 4);
    _Float16* h1   = (_Float16*)alloc(M * DFF * 2);

    // weights -> f16 transposed
    cvt_t_kernel<<<(DD * DD + 255) / 256, 256, 0, stream>>>(Wq, Wq_t, DD, DD);
    cvt_t_kernel<<<(DD * DD + 255) / 256, 256, 0, stream>>>(Wk, Wk_t, DD, DD);
    cvt_t_kernel<<<(DD * DD + 255) / 256, 256, 0, stream>>>(Wv, Wv_t, DD, DD);
    cvt_t_kernel<<<(DD * DD + 255) / 256, 256, 0, stream>>>(Wo, Wo_t, DD, DD);
    cvt_t_kernel<<<(DD * DFF + 255) / 256, 256, 0, stream>>>(W1, W1_t, DD, DFF);
    cvt_t_kernel<<<(DFF * DD + 255) / 256, 256, 0, stream>>>(W2, W2_t, DFF, DD);

    // xn1 = layernorm(x)
    layernorm_kernel<<<M, 256, 0, stream>>>(x, xn);

    // Q,K,V projections (WMMA)
    dim3 gqkv(DD / 64, M / 64);
    gemm_f16_kernel<<<gqkv, 128, 0, stream>>>(xn, Wq_t, bq, nullptr, qb,  (int)M, DD, DD, 0);
    gemm_f16_kernel<<<gqkv, 128, 0, stream>>>(xn, Wk_t, bk, nullptr, kb,  (int)M, DD, DD, 0);
    gemm_f16_kernel<<<gqkv, 128, 0, stream>>>(xn, Wv_t, bv, nullptr, vtb, (int)M, DD, DD, 1);

    // flash attention -> attn (token-major f16)
    flash_attn_kernel<<<dim3(BB * HH, NN / 64), 128, 0, stream>>>(qb, kb, vtb, attn);

    // x_skip = attn @ Wo + bo + x
    gemm_f16_kernel<<<gqkv, 128, 0, stream>>>(attn, Wo_t, bo, x, xsk, (int)M, DD, DD, 2);

    // xn2 = layernorm(x_skip)  (reuse xn buffer)
    layernorm_kernel<<<M, 256, 0, stream>>>(xsk, xn);

    // h1 = gelu(xn2 @ W1 + b1)
    gemm_f16_kernel<<<dim3(DFF / 64, M / 64), 128, 0, stream>>>(xn, W1_t, b1, nullptr, h1,
                                                               (int)M, DD, DFF, 3);
    // out = h1 @ W2 + b2 + x_skip
    gemm_f16_kernel<<<gqkv, 128, 0, stream>>>(h1, W2_t, b2, xsk, out, (int)M, DFF, DD, 4);
}
